// PerformerNodeClassifier_42082089566414
// MI455X (gfx1250) — matrix operand order repaired and checked by reading, not compiled, not run on hardware
//
#include <hip/hip_runtime.h>
#include <math.h>

// ---------------------------------------------------------------------------
// PerformerNodeClassifier on gfx1250 (MI455X): bf16 WMMA GEMMs + fp32 accum.
// ---------------------------------------------------------------------------

typedef __attribute__((ext_vector_type(16))) __bf16 v16bf;
typedef __attribute__((ext_vector_type(8)))  float  v8f;
typedef unsigned short ushort_t;

constexpr int kN  = 65536;
constexpr int kIN = 512;
constexpr int kE  = 128;
constexpr int kH  = 4;
constexpr int kDH = 32;
constexpr int kM  = 256;
constexpr int kL  = 2;

#define DEVFN __device__ __forceinline__

DEVFN ushort_t f2bf(float f) {
  unsigned u = __float_as_uint(f);
  u += 0x7FFFu + ((u >> 16) & 1u);           // round-to-nearest-even
  return (ushort_t)(u >> 16);
}
DEVFN float bf2f(ushort_t h) { return __uint_as_float(((unsigned)h) << 16); }

// order-preserving float <-> uint mapping for atomicMax on floats
DEVFN unsigned fmapu(float f) {
  unsigned u = __float_as_uint(f);
  return (u & 0x80000000u) ? ~u : (u | 0x80000000u);
}
DEVFN float fumap(unsigned u) {
  return (u & 0x80000000u) ? __uint_as_float(u & 0x7fffffffu) : __uint_as_float(~u);
}

// ---------------------------------------------------------------------------
// Generic bf16 WMMA GEMM:  O = epilogue( A[rows,K] @ B[K,cols] )
//   grid = (rows/128, cols/(16*NT)), block = 256 (8 waves).
//   Each wave computes a 16 x (16*NT) strip: the A fragment is loaded once per
//   32-wide K step and reused for NT WMMAs (B fragments from LDS B^T tile).
//   K is staged in chunks of 256 to bound LDS (16*NT cols * 264 * 2B).
//   ACT: 0 = none, 1 = exact GELU, 2 = ReLU.  Epilogue: +bias, act, +R (f32),
//   write Of (f32) and/or Ob (bf16).
// ---------------------------------------------------------------------------
template <int ACT, int NT>
__global__ __launch_bounds__(256)
void gemm_wmma(const ushort_t* __restrict__ A, int lda,
               const ushort_t* __restrict__ B, int ldb, int K,
               const float* __restrict__ bias,
               const float* __restrict__ R, int ldr,
               float* __restrict__ Of, ushort_t* __restrict__ Ob, int ldc)
{
  constexpr int BCOLS = 16 * NT;
  __shared__ ushort_t bT[BCOLS * (256 + 8)];   // B^T chunk, padded stride
  const int tid  = threadIdx.x;
  const int wave = tid >> 5;
  const int lane = tid & 31;
  const int col0 = blockIdx.y * BCOLS;
  const int l16  = lane & 15;
  const int half = lane >> 4;
  const int rowBase = ((int)blockIdx.x * 8 + wave) << 4;
  const int aoff = half << 3;                  // A K sub-offset 0 / 8
  const ushort_t* aRow = A + (size_t)(rowBase + l16) * lda;

  union F { v16bf v; uint4 q[2]; };
  v8f acc[NT];
#pragma unroll
  for (int t = 0; t < NT; ++t) acc[t] = {};

  for (int k0 = 0; k0 < K; k0 += 256) {
    const int KC = (K - k0) < 256 ? (K - k0) : 256;
    const int sT = KC + 8;                     // padded LDS stride (mult of 8)
    if (k0) __syncthreads();
    // stage B^T chunk (coalesced global reads, strided LDS writes)
    for (int idx = tid; idx < KC * BCOLS; idx += 256) {
      const int k = idx / BCOLS, c = idx % BCOLS;
      bT[c * sT + k] = B[(size_t)(k0 + k) * ldb + col0 + c];
    }
    __syncthreads();

    for (int kk = 0; kk < KC; kk += 32) {
      // A fragment: lanes 0-15 hold K = +0..7 & +16..23 ; lanes 16-31: +8
      F af;
      af.q[0] = *(const uint4*)(aRow + k0 + kk + aoff);
      af.q[1] = *(const uint4*)(aRow + k0 + kk + 16 + aoff);
      if (kk + 64 < KC) __builtin_prefetch(aRow + k0 + kk + 64, 0, 1);
#pragma unroll
      for (int t = 0; t < NT; ++t) {
        // B fragment: lanes 0-15 -> K +0..7 & +16..23 of column l16; lanes 16-31: +8
        F bf;
        const ushort_t* bc = &bT[(t * 16 + l16) * sT + (half << 3) + kk];
        bf.q[0] = *(const uint4*)(bc);
        bf.q[1] = *(const uint4*)(bc + 16);
        acc[t] = __builtin_amdgcn_wmma_f32_16x16x32_bf16(
            false, af.v, false, bf.v, (short)0, acc[t], false, false);
      }
    }
  }

  // C/D layout: lanes 0-15 -> rows rowBase+0..7, lanes 16-31 -> rowBase+8..15
#pragma unroll
  for (int t = 0; t < NT; ++t) {
    const int col = col0 + t * 16 + l16;
    const float bval = bias ? bias[col] : 0.f;
#pragma unroll
    for (int i = 0; i < 8; ++i) {
      float x = acc[t][i] + bval;
      if (ACT == 1) x = 0.5f * x * (1.f + erff(x * 0.70710678118f));
      if (ACT == 2) x = fmaxf(x, 0.f);
      const size_t row = (size_t)(rowBase + (half << 3) + i);
      if (R)  x += R[row * ldr + col];
      if (Of) Of[row * ldc + col] = x;
      if (Ob) Ob[row * ldc + col] = f2bf(x);
    }
  }
}

// ---------------------------------------------------------------------------
// LayerNorm over E=128, one wave per row
// ---------------------------------------------------------------------------
__global__ __launch_bounds__(256)
void layernorm_k(const float* __restrict__ X, const float* __restrict__ g,
                 const float* __restrict__ b, float* __restrict__ Of,
                 ushort_t* __restrict__ Ob)
{
  const size_t row = (size_t)blockIdx.x * 8 + (threadIdx.x >> 5);
  const int lane = threadIdx.x & 31;
  const float4 v = *(const float4*)(X + row * kE + lane * 4);
  float s = v.x + v.y + v.z + v.w;
  for (int m = 16; m; m >>= 1) s += __shfl_xor(s, m, 32);
  const float mu = s * (1.f / kE);
  float d[4] = {v.x - mu, v.y - mu, v.z - mu, v.w - mu};
  float s2 = d[0]*d[0] + d[1]*d[1] + d[2]*d[2] + d[3]*d[3];
  for (int m = 16; m; m >>= 1) s2 += __shfl_xor(s2, m, 32);
  const float rs = rsqrtf(s2 * (1.f / kE) + 1e-5f);
#pragma unroll
  for (int j = 0; j < 4; ++j) {
    const int e = lane * 4 + j;
    const float o = d[j] * rs * g[e] + b[e];
    if (Of) Of[row * kE + e] = o;
    if (Ob) Ob[row * kE + e] = f2bf(o);
  }
}

// ---------------------------------------------------------------------------
// FAVOR+ prep: Qs = bf16(q * dn); diag[n,h] = 0.5 * sum_d (q*dn)^2 per head
// ---------------------------------------------------------------------------
__global__ __launch_bounds__(256)
void featprep_k(const float* __restrict__ Q, ushort_t* __restrict__ Qs,
                float* __restrict__ diag)
{
  const size_t row = (size_t)blockIdx.x * 8 + (threadIdx.x >> 5);
  const int lane = threadIdx.x & 31;
  const float dn = 0.42044820763f;            // 32^-0.25
  const float4 v = *(const float4*)(Q + row * kE + lane * 4);
  float q[4] = {v.x * dn, v.y * dn, v.z * dn, v.w * dn};
  float ss = q[0]*q[0] + q[1]*q[1] + q[2]*q[2] + q[3]*q[3];
  ss += __shfl_xor(ss, 1, 32);
  ss += __shfl_xor(ss, 2, 32);
  ss += __shfl_xor(ss, 4, 32);                // reduce within 8-lane head group
#pragma unroll
  for (int j = 0; j < 4; ++j) Qs[row * kE + lane * 4 + j] = f2bf(q[j]);
  if ((lane & 7) == 0) diag[row * kH + (lane >> 3)] = 0.5f * ss;
}

// row max over M=256, one wave per row (rows enumerated as h*N+n)
__global__ __launch_bounds__(256)
void rowmax_k(const float* __restrict__ X, float* __restrict__ out)
{
  const size_t row = (size_t)blockIdx.x * 8 + (threadIdx.x >> 5);
  const int lane = threadIdx.x & 31;
  const float* p = X + row * kM;
  float mx = -3.4e38f;
#pragma unroll
  for (int j = 0; j < 8; ++j) mx = fmaxf(mx, p[lane * 8 + j]);
  for (int m = 16; m; m >>= 1) mx = fmaxf(mx, __shfl_xor(mx, m, 32));
  if (lane == 0) out[row] = mx;
}

__global__ void initmax_k(unsigned* out) { *out = fmapu(-3.4e38f); }

__global__ __launch_bounds__(256)
void gmax_k(const float* __restrict__ X, size_t count, unsigned* __restrict__ out)
{
  __shared__ float sm[8];
  float mx = -3.4e38f;
  for (size_t i = (size_t)blockIdx.x * 256 + threadIdx.x; i < count;
       i += (size_t)gridDim.x * 256)
    mx = fmaxf(mx, X[i]);
  for (int m = 16; m; m >>= 1) mx = fmaxf(mx, __shfl_xor(mx, m, 32));
  if ((threadIdx.x & 31) == 0) sm[threadIdx.x >> 5] = mx;
  __syncthreads();
  if (threadIdx.x == 0) {
    float bmx = sm[0];
    for (int w = 1; w < 8; ++w) bmx = fmaxf(bmx, sm[w]);
    atomicMax(out, fmapu(bmx));
  }
}

// qf = ratio * (exp(dd - diag - stab_row) + eps), bf16 out
__global__ __launch_bounds__(256)
void expq_k(const float* __restrict__ dd, const float* __restrict__ diag,
            const float* __restrict__ stab, ushort_t* __restrict__ out)
{
  const size_t total = (size_t)kH * kN * kM;
  for (size_t i = (size_t)blockIdx.x * 256 + threadIdx.x; i < total;
       i += (size_t)gridDim.x * 256) {
    const int h = (int)(i >> 24);
    const int n = (int)((i >> 8) & (kN - 1));
    const float v = 0.0625f *
        (expf(dd[i] - diag[(size_t)n * kH + h] - stab[(size_t)h * kN + n]) + 1e-4f);
    out[i] = f2bf(v);
  }
}

__global__ __launch_bounds__(256)
void expk_k(const float* __restrict__ dd, const float* __restrict__ diag,
            const unsigned* __restrict__ stab_u, ushort_t* __restrict__ out)
{
  const float st = fumap(*stab_u);
  const size_t total = (size_t)kH * kN * kM;
  for (size_t i = (size_t)blockIdx.x * 256 + threadIdx.x; i < total;
       i += (size_t)gridDim.x * 256) {
    const int h = (int)(i >> 24);
    const int n = (int)((i >> 8) & (kN - 1));
    const float v = 0.0625f *
        (expf(dd[i] - diag[(size_t)n * kH + h] - st) + 1e-4f);
    out[i] = f2bf(v);
  }
}

__global__ void zero_k(float* p, int count) {
  const int i = blockIdx.x * 256 + threadIdx.x;
  if (i < count) p[i] = 0.f;
}

// kfsum[h,m] = sum_n kf[h,n,m]   (grid = (N/1024, H), thread = m)
__global__ __launch_bounds__(256)
void kfsum_k(const ushort_t* __restrict__ kf, float* __restrict__ ksum)
{
  const int h = blockIdx.y, m = threadIdx.x;
  const int n0 = blockIdx.x * 1024;
  const ushort_t* base = kf + ((size_t)h * kN + n0) * kM + m;
  float s = 0.f;
  for (int n = 0; n < 1024; ++n) s += bf2f(base[(size_t)n * kM]);
  atomicAdd(&ksum[h * kM + m], s);
}

// ctx[h,m,d] = sum_n kf[h,n,m] * v[n, h*32+d]   (grid = (N/2048, H))
// V tile is staged into LDS with GLOBAL_LOAD_ASYNC_TO_LDS_B32 (ASYNCcnt).
__global__ __launch_bounds__(256)
void ctx_k(const ushort_t* __restrict__ kf, const float* __restrict__ V,
           float* __restrict__ ctx)
{
  __shared__ float vt[8 * kDH];            // at LDS offset 0 (only LDS object)
  const int h = blockIdx.y, m = threadIdx.x;
  const int n0 = blockIdx.x * 2048;
  const int r = threadIdx.x >> 5, d = threadIdx.x & 31;
  const unsigned ldsOff = (unsigned)threadIdx.x * 4u;   // vt[r*32 + d]
  float acc[kDH];
#pragma unroll
  for (int j = 0; j < kDH; ++j) acc[j] = 0.f;

  for (int nb = 0; nb < 2048; nb += 8) {
    const float* gp = V + (size_t)(n0 + nb + r) * kE + h * kDH + d;
    // async copy one dword per lane: LDS[vt + tid*4] = *gp   (GV mode)
    asm volatile("global_load_async_to_lds_b32 %0, %1, off"
                 :: "v"(ldsOff), "v"((unsigned long long)(size_t)gp)
                 : "memory");
    asm volatile("s_wait_asynccnt 0x0" ::: "memory");
    __syncthreads();
#pragma unroll
    for (int rr = 0; rr < 8; ++rr) {
      const float kv = bf2f(kf[((size_t)h * kN + n0 + nb + rr) * kM + m]);
#pragma unroll
      for (int j = 0; j < kDH; ++j) acc[j] += kv * vt[rr * kDH + j];
    }
    __syncthreads();
  }
  float* cp = ctx + ((size_t)h * kM + m) * kDH;
#pragma unroll
  for (int j = 0; j < kDH; ++j) atomicAdd(&cp[j], acc[j]);
}

// Dinv[h,n] = 1 / dot(qf[h,n,:], kfsum[h,:])  (wave per (h,n))
__global__ __launch_bounds__(256)
void dinv_k(const ushort_t* __restrict__ qf, const float* __restrict__ ksum,
            float* __restrict__ Dinv)
{
  const size_t r = (size_t)blockIdx.x * 8 + (threadIdx.x >> 5);  // h*N + n
  const int lane = threadIdx.x & 31;
  const int h = (int)(r >> 16);
  const ushort_t* p = qf + r * kM;
  const float* ks = ksum + h * kM;
  float s = 0.f;
#pragma unroll
  for (int j = 0; j < 8; ++j) {
    const int m = lane * 8 + j;
    s += bf2f(p[m]) * ks[m];
  }
  for (int mk = 16; mk; mk >>= 1) s += __shfl_xor(s, mk, 32);
  if (lane == 0) Dinv[r] = 1.f / s;
}

// a_bf16[n,e] = bf16( a_f32[n,e] * Dinv[e/32, n] )
__global__ __launch_bounds__(256)
void ascale_k(const float* __restrict__ A, const float* __restrict__ Dinv,
              ushort_t* __restrict__ Ab)
{
  const size_t total = (size_t)kN * kE;
  for (size_t i = (size_t)blockIdx.x * 256 + threadIdx.x; i < total;
       i += (size_t)gridDim.x * 256) {
    const int e = (int)(i & (kE - 1));
    const int n = (int)(i >> 7);
    Ab[i] = f2bf(A[i] * Dinv[(size_t)(e >> 5) * kN + n]);
  }
}

__global__ __launch_bounds__(256)
void cvt_k(const float* __restrict__ in, ushort_t* __restrict__ out, size_t count)
{
  for (size_t i = (size_t)blockIdx.x * 256 + threadIdx.x; i < count;
       i += (size_t)gridDim.x * 256)
    out[i] = f2bf(in[i]);
}

// out[c*R + r] = bf16(in[r*C + c])  (proj -> projT)
__global__ void tcvt_k(const float* __restrict__ in, ushort_t* __restrict__ out,
                       int Rr, int Cc)
{
  const int i = blockIdx.x * 256 + threadIdx.x;
  if (i < Rr * Cc) {
    const int r = i / Cc, c = i % Cc;
    out[(size_t)c * Rr + r] = f2bf(in[i]);
  }
}

// sigmoid classifier: out[n] = sigmoid(h[n,:]·Wc + bc)
__global__ __launch_bounds__(256)
void cls_k(const float* __restrict__ Hm, const float* __restrict__ Wc,
           const float* __restrict__ bc, float* __restrict__ out)
{
  const size_t row = (size_t)blockIdx.x * 8 + (threadIdx.x >> 5);
  const int lane = threadIdx.x & 31;
  const float4 v = *(const float4*)(Hm + row * kE + lane * 4);
  const float4 w = *(const float4*)(Wc + lane * 4);
  float s = v.x * w.x + v.y * w.y + v.z * w.z + v.w * w.w;
  for (int m = 16; m; m >>= 1) s += __shfl_xor(s, m, 32);
  if (lane == 0) out[row] = 1.f / (1.f + expf(-(s + bc[0])));
}

// ---------------------------------------------------------------------------
extern "C" void kernel_launch(void* const* d_in, const int* in_sizes, int n_in,
                              void* d_out, int out_size, void* d_ws, size_t ws_size,
                              hipStream_t stream)
{
  (void)in_sizes; (void)n_in; (void)out_size;

  const float* x    = (const float*)d_in[0];
  const float* Wpin = (const float*)d_in[1];
  const float* bpin = (const float*)d_in[2];
  const float* lag  = (const float*)d_in[3];
  const float* lab  = (const float*)d_in[4];
  const float* Wq   = (const float*)d_in[5];
  const float* Wk   = (const float*)d_in[6];
  const float* Wv   = (const float*)d_in[7];
  const float* Wo   = (const float*)d_in[8];
  const float* bo   = (const float*)d_in[9];
  const float* lbg  = (const float*)d_in[10];
  const float* lbb  = (const float*)d_in[11];
  const float* Wf1  = (const float*)d_in[12];
  const float* bf1  = (const float*)d_in[13];
  const float* Wf2  = (const float*)d_in[14];
  const float* bf2  = (const float*)d_in[15];
  const float* Wpo  = (const float*)d_in[16];
  const float* bpo  = (const float*)d_in[17];
  const float* n1g  = (const float*)d_in[18];
  const float* n1b  = (const float*)d_in[19];
  const float* Wn1  = (const float*)d_in[20];
  const float* bn1  = (const float*)d_in[21];
  const float* Wn2  = (const float*)d_in[22];
  const float* bn2  = (const float*)d_in[23];
  const float* n2g  = (const float*)d_in[24];
  const float* n2b  = (const float*)d_in[25];
  const float* Wc   = (const float*)d_in[26];
  const float* bc   = (const float*)d_in[27];
  const float* proj = (const float*)d_in[28];

  // ---- workspace bump allocator (256B aligned) ----
  char* base = (char*)d_ws;
  size_t off = 0;
  auto alloc = [&](size_t bytes) -> char* {
    char* p = base + off;
    off = (off + bytes + 255) & ~(size_t)255;
    return p;
  };

  ushort_t* x_b    = (ushort_t*)alloc((size_t)kN * kIN * 2);
  ushort_t* wpin_b = (ushort_t*)alloc((size_t)kIN * kE * 2);
  ushort_t* wq_b   = (ushort_t*)alloc((size_t)kL * kE * kE * 2);
  ushort_t* wk_b   = (ushort_t*)alloc((size_t)kL * kE * kE * 2);
  ushort_t* wv_b   = (ushort_t*)alloc((size_t)kL * kE * kE * 2);
  ushort_t* wo_b   = (ushort_t*)alloc((size_t)kL * kE * kE * 2);
  ushort_t* wpo_b  = (ushort_t*)alloc((size_t)kL * kE * kE * 2);
  ushort_t* wf1_b  = (ushort_t*)alloc((size_t)kL * kE * 4 * kE * 2);
  ushort_t* wf2_b  = (ushort_t*)alloc((size_t)kL * 4 * kE * kE * 2);
  ushort_t* wn1_b  = (ushort_t*)alloc((size_t)kL * kE * 4 * kE * 2);
  ushort_t* wn2_b  = (ushort_t*)alloc((size_t)kL * 4 * kE * kE * 2);
  ushort_t* pjt_b  = (ushort_t*)alloc((size_t)kL * kDH * kM * 2);

  float*    h_f    = (float*)alloc((size_t)kN * kE * 4);
  ushort_t* h_b    = (ushort_t*)alloc((size_t)kN * kE * 2);
  ushort_t* y_b    = (ushort_t*)alloc((size_t)kN * kE * 2);
  float*    q_f    = (float*)alloc((size_t)kN * kE * 4);
  float*    k_f    = (float*)alloc((size_t)kN * kE * 4);
  float*    v_f    = (float*)alloc((size_t)kN * kE * 4);
  ushort_t* qs_b   = (ushort_t*)alloc((size_t)kN * kE * 2);
  ushort_t* ks_b   = (ushort_t*)alloc((size_t)kN * kE * 2);
  float*    diag_q = (float*)alloc((size_t)kN * kH * 4);
  float*    diag_k = (float*)alloc((size_t)kN * kH * 4);
  float*    dd     = (float*)alloc((size_t)kH * kN * kM * 4);
  ushort_t* qf_b   = (ushort_t*)alloc((size_t)kH * kN * kM * 2);
  ushort_t* kf_b   = (ushort_t*)alloc((size_t)kH * kN * kM * 2);
  float*    qstab  = (float*)alloc((size_t)kH * kN * 4);
  unsigned* kstab  = (unsigned*)alloc(256);
  float*    ksum   = (float*)alloc((size_t)kH * kM * 4);
  float*    ctxf   = (float*)alloc((size_t)kH * kM * kDH * 4);
  ushort_t* ctx_b  = (ushort_t*)alloc((size_t)kH * kM * kDH * 2);
  float*    dinv   = (float*)alloc((size_t)kH * kN * 4);
  float*    a_f    = (float*)alloc((size_t)kN * kE * 4);
  ushort_t* a_b    = (ushort_t*)alloc((size_t)kN * kE * 2);
  float*    h1_f   = (float*)alloc((size_t)kN * kE * 4);
  ushort_t* y2_b   = (ushort_t*)alloc((size_t)kN * kE * 2);
  ushort_t* big_b  = (ushort_t*)alloc((size_t)kN * 4 * kE * 2);  // ff / relu
  float*    h2_f   = (float*)alloc((size_t)kN * kE * 4);
  ushort_t* h2_b   = (ushort_t*)alloc((size_t)kN * kE * 2);
  float*    t_f    = (float*)alloc((size_t)kN * kE * 4);

  if (off > ws_size) return;   // insufficient scratch; bail deterministically

  auto cvt = [&](const float* s, ushort_t* dptr, size_t cnt) {
    cvt_k<<<1024, 256, 0, stream>>>(s, dptr, cnt);
  };
  auto gemm = [&](int cols, int K, const ushort_t* A, int lda, const ushort_t* B,
                  const float* bias, int act, const float* R, int ldr,
                  float* Of, ushort_t* Ob, int ldc) {
    if ((cols & 63) == 0) {
      dim3 g(kN / 128, cols / 64);
      if (act == 1)
        gemm_wmma<1, 4><<<g, 256, 0, stream>>>(A, lda, B, cols, K, bias, R, ldr, Of, Ob, ldc);
      else if (act == 2)
        gemm_wmma<2, 4><<<g, 256, 0, stream>>>(A, lda, B, cols, K, bias, R, ldr, Of, Ob, ldc);
      else
        gemm_wmma<0, 4><<<g, 256, 0, stream>>>(A, lda, B, cols, K, bias, R, ldr, Of, Ob, ldc);
    } else {                    // cols == 32 (per-head a-GEMM), act == 0
      dim3 g(kN / 128, cols / 32);
      gemm_wmma<0, 2><<<g, 256, 0, stream>>>(A, lda, B, cols, K, bias, R, ldr, Of, Ob, ldc);
    }
  };

  // ---- one-time (per call) bf16 conversions ----
  cvt(x, x_b, (size_t)kN * kIN);
  cvt(Wpin, wpin_b, (size_t)kIN * kE);
  cvt(Wq, wq_b, (size_t)kL * kE * kE);
  cvt(Wk, wk_b, (size_t)kL * kE * kE);
  cvt(Wv, wv_b, (size_t)kL * kE * kE);
  cvt(Wo, wo_b, (size_t)kL * kE * kE);
  cvt(Wpo, wpo_b, (size_t)kL * kE * kE);
  cvt(Wf1, wf1_b, (size_t)kL * kE * 4 * kE);
  cvt(Wf2, wf2_b, (size_t)kL * 4 * kE * kE);
  cvt(Wn1, wn1_b, (size_t)kL * kE * 4 * kE);
  cvt(Wn2, wn2_b, (size_t)kL * 4 * kE * kE);
  for (int l = 0; l < kL; ++l)
    tcvt_k<<<(kM * kDH + 255) / 256, 256, 0, stream>>>(
        proj + (size_t)l * kM * kDH, pjt_b + (size_t)l * kDH * kM, kM, kDH);

  // ---- input projection: h = x @ Wpin + bpin ----
  gemm(kE, kIN, x_b, kIN, wpin_b, bpin, 0, nullptr, 0, h_f, nullptr, kE);

  const int RW = kN / 8;          // wave-per-row grids
  const size_t DDCNT = (size_t)kH * kN * kM;

  for (int l = 0; l < kL; ++l) {
    const ushort_t* wq_l  = wq_b  + (size_t)l * kE * kE;
    const ushort_t* wk_l  = wk_b  + (size_t)l * kE * kE;
    const ushort_t* wv_l  = wv_b  + (size_t)l * kE * kE;
    const ushort_t* wo_l  = wo_b  + (size_t)l * kE * kE;
    const ushort_t* wpo_l = wpo_b + (size_t)l * kE * kE;
    const ushort_t* wf1_l = wf1_b + (size_t)l * kE * 4 * kE;
    const ushort_t* wf2_l = wf2_b + (size_t)l * 4 * kE * kE;
    const ushort_t* wn1_l = wn1_b + (size_t)l * kE * 4 * kE;
    const ushort_t* wn2_l = wn2_b + (size_t)l * 4 * kE * kE;
    const ushort_t* pjt_l = pjt_b + (size_t)l * kDH * kM;

    // --- inner Performer: pre-LN attention ---
    layernorm_k<<<RW, 256, 0, stream>>>(h_f, lag + l * kE, lab + l * kE,
                                        nullptr, y_b);
    gemm(kE, kE, y_b, kE, wq_l, nullptr, 0, nullptr, 0, q_f, nullptr, kE);
    gemm(kE, kE, y_b, kE, wk_l, nullptr, 0, nullptr, 0, k_f, nullptr, kE);
    gemm(kE, kE, y_b, kE, wv_l, nullptr, 0, nullptr, 0, v_f, nullptr, kE);

    featprep_k<<<RW, 256, 0, stream>>>(q_f, qs_b, diag_q);
    featprep_k<<<RW, 256, 0, stream>>>(k_f, ks_b, diag_k);

    // dd_q = (q*dn) @ proj^T per head, then qf = exp(dd - diag - rowmax)
    for (int h = 0; h < kH; ++h)
      gemm(kM, kDH, qs_b + h * kDH, kE, pjt_l, nullptr, 0, nullptr, 0,
           dd + (size_t)h * kN * kM, nullptr, kM);
    rowmax_k<<<(kH * kN) / 8, 256, 0, stream>>>(dd, qstab);
    expq_k<<<32768, 256, 0, stream>>>(dd, diag_q, qstab, qf_b);

    // dd_k, global max, kf
    for (int h = 0; h < kH; ++h)
      gemm(kM, kDH, ks_b + h * kDH, kE, pjt_l, nullptr, 0, nullptr, 0,
           dd + (size_t)h * kN * kM, nullptr, kM);
    initmax_k<<<1, 1, 0, stream>>>(kstab);
    gmax_k<<<4096, 256, 0, stream>>>(dd, DDCNT, kstab);
    expk_k<<<32768, 256, 0, stream>>>(dd, diag_k, kstab, kf_b);

    // kfsum, ctx = kf^T @ v, D^-1
    zero_k<<<(kH * kM + 255) / 256, 256, 0, stream>>>(ksum, kH * kM);
    kfsum_k<<<dim3(kN / 1024, kH), 256, 0, stream>>>(kf_b, ksum);
    zero_k<<<(kH * kM * kDH + 255) / 256, 256, 0, stream>>>(ctxf, kH * kM * kDH);
    ctx_k<<<dim3(kN / 2048, kH), 256, 0, stream>>>(kf_b, v_f, ctxf);
    cvt(ctxf, ctx_b, (size_t)kH * kM * kDH);
    dinv_k<<<(kH * kN) / 8, 256, 0, stream>>>(qf_b, ksum, dinv);

    // a = (qf @ ctx) * D^-1  -> bf16
    for (int h = 0; h < kH; ++h)
      gemm(kDH, kM, qf_b + (size_t)h * kN * kM, kM, ctx_b + h * kM * kDH,
           nullptr, 0, nullptr, 0, a_f + h * kDH, nullptr, kE);
    ascale_k<<<8192, 256, 0, stream>>>(a_f, dinv, a_b);

    // h1 = h + a @ Wo + bo ; inner pre-LN GELU FFN
    gemm(kE, kE, a_b, kE, wo_l, bo + l * kE, 0, h_f, kE, h1_f, nullptr, kE);
    layernorm_k<<<RW, 256, 0, stream>>>(h1_f, lbg + l * kE, lbb + l * kE,
                                        nullptr, y2_b);
    gemm(4 * kE, kE, y2_b, kE, wf1_l, bf1 + l * 4 * kE, 1, nullptr, 0,
         nullptr, big_b, 4 * kE);
    gemm(kE, 4 * kE, big_b, 4 * kE, wf2_l, bf2 + l * kE, 0, h1_f, kE,
         h2_f, h2_b, kE);

    // outer block: post-LN residual attn + post-LN ReLU FFN
    gemm(kE, kE, h2_b, kE, wpo_l, bpo + l * kE, 0, h_f, kE, t_f, nullptr, kE);
    layernorm_k<<<RW, 256, 0, stream>>>(t_f, n1g + l * kE, n1b + l * kE,
                                        h_f, h_b);
    gemm(4 * kE, kE, h_b, kE, wn1_l, bn1 + l * 4 * kE, 2, nullptr, 0,
         nullptr, big_b, 4 * kE);
    gemm(kE, 4 * kE, big_b, 4 * kE, wn2_l, bn2 + l * kE, 0, h_f, kE,
         t_f, nullptr, kE);
    layernorm_k<<<RW, 256, 0, stream>>>(t_f, n2g + l * kE, n2b + l * kE,
                                        h_f, h_b);
  }

  // classifier
  cls_k<<<RW, 256, 0, stream>>>(h_f, Wc, bc, (float*)d_out);
}
